// VectorQuantizer_57466662420906
// MI455X (gfx1250) — matrix operand order repaired and checked by reading, not compiled, not run on hardware
//
#include <hip/hip_runtime.h>
#include <hip/hip_bf16.h>

// ---------------------------------------------------------------------------
// VQ-VAE vector quantizer for MI455X (gfx1250, wave32, WMMA).
// z: [16,128,64,64] f32, E: [1024,128] f32.
// Outputs (concat in d_out as f32): quantized [16,128,64,64], vq_loss [1],
// indices [65536].
//
// Round 3: B fragments staged in distinct registers ahead of the WMMA chain
// so ds_load latency overlaps the v_wmma chain (staggered s_wait_dscnt
// instead of wait-0 before every WMMA); en2 load hoisted to tile start.
// Codebook streams through a 2x32KB LDS double buffer via
// GLOBAL_LOAD_ASYNC_TO_LDS_B128 (ASYNCcnt).
// ---------------------------------------------------------------------------

typedef __bf16 v4bf  __attribute__((ext_vector_type(4)));
typedef __bf16 v8bf  __attribute__((ext_vector_type(8)));
typedef __bf16 v16bf __attribute__((ext_vector_type(16)));
typedef float  v8f   __attribute__((ext_vector_type(8)));

#define N_TOK   65536      // 16*64*64
#define C_DIM   128
#define K_CODES 1024
#define HW      4096       // 64*64
#define OUT_Q   8388608    // 16*128*64*64
#define LOSS_OFF OUT_Q
#define IDX_OFF (OUT_Q + 1)

// ---------------------------------------------------------------------------
// Kernel 1: transpose z [B,C,HW] -> z_bf16 [N=B*HW, C] (row-major bf16)
// ---------------------------------------------------------------------------
__global__ __launch_bounds__(256) void vq_zprep(const float* __restrict__ z,
                                                __bf16* __restrict__ zb) {
  __shared__ float tile[32][33];
  const int hw0 = blockIdx.x * 32;
  const int c0  = blockIdx.y * 32;
  const int b   = blockIdx.z;
  const int tx  = threadIdx.x;      // 0..31
  const int ty  = threadIdx.y;      // 0..7
  const float* zp = z + (size_t)b * C_DIM * HW;
#pragma unroll
  for (int i = 0; i < 4; ++i) {
    int c = c0 + ty + i * 8;
    tile[ty + i * 8][tx] = zp[(size_t)c * HW + hw0 + tx];
  }
  __syncthreads();
#pragma unroll
  for (int i = 0; i < 4; ++i) {
    int hwl = ty + i * 8;
    size_t n = (size_t)b * HW + hw0 + hwl;
    zb[n * C_DIM + c0 + tx] = (__bf16)tile[tx][hwl];
  }
}

// ---------------------------------------------------------------------------
// Kernel 2: codebook -> bf16, per-code squared norm; zero the loss accum.
// One wave per code row (32 lanes x 4 floats = 128).
// ---------------------------------------------------------------------------
__global__ __launch_bounds__(256) void vq_eprep(const float* __restrict__ emb,
                                                __bf16* __restrict__ eb,
                                                float* __restrict__ en2,
                                                double* __restrict__ lossAcc) {
  const int tid  = threadIdx.x;
  const int lane = tid & 31;
  const int wave = tid >> 5;
  const int row  = blockIdx.x * 8 + wave;   // 128 blocks * 8 = 1024 rows
  const float4 v = ((const float4*)(emb + (size_t)row * C_DIM))[lane];
  __bf16 b0 = (__bf16)v.x, b1 = (__bf16)v.y, b2 = (__bf16)v.z, b3 = (__bf16)v.w;
  v4bf pk = {b0, b1, b2, b3};
  *(v4bf*)(eb + (size_t)row * C_DIM + lane * 4) = pk;
  float f0 = (float)b0, f1 = (float)b1, f2 = (float)b2, f3 = (float)b3;
  float s = f0 * f0 + f1 * f1 + f2 * f2 + f3 * f3;
#pragma unroll
  for (int m = 16; m >= 1; m >>= 1) s += __shfl_xor(s, m, 32);
  if (lane == 0) en2[row] = s;
  if (blockIdx.x == 0 && tid == 0) *lossAcc = 0.0;
}

// ---------------------------------------------------------------------------
// Async copy of one 32 KB codebook chunk (128 codes x 128 bf16) to LDS.
// Each of the 256 threads issues 8 async b128 transfers (IOFFSET applies to
// both the global and the LDS address, so one addr pair covers all 8).
// Tracked with ASYNCcnt; no VGPR staging, no dscnt stall.
// ---------------------------------------------------------------------------
__device__ __forceinline__ void vq_async_chunk(const __bf16* __restrict__ gsrc,
                                               unsigned lds_base, int tid) {
  const char* gptr = (const char*)gsrc + tid * 16;
  unsigned lo = lds_base + tid * 16;
#define VQ_ALD(OFF)                                                       \
  asm volatile("global_load_async_to_lds_b128 %0, %1, off offset:" #OFF   \
               :: "v"(lo), "v"(gptr) : "memory")
  VQ_ALD(0);     VQ_ALD(4096);  VQ_ALD(8192);  VQ_ALD(12288);
  VQ_ALD(16384); VQ_ALD(20480); VQ_ALD(24576); VQ_ALD(28672);
#undef VQ_ALD
}

// ---------------------------------------------------------------------------
// Kernel 3: argmin via bf16 WMMA, double-buffered async codebook streaming.
// 8 waves/block; wave owns 16 z-rows (A tile in regs). score = ||e||^2 -
// 2*z.e (||z||^2 is row-constant and drops out of argmin).
// ---------------------------------------------------------------------------
__global__ __launch_bounds__(256) void vq_argmin(const __bf16* __restrict__ zb,
                                                 const __bf16* __restrict__ eb,
                                                 const float* __restrict__ en2,
                                                 int* __restrict__ idx) {
  __shared__ __align__(16) __bf16 lds_e[2][128 * C_DIM];  // 2 x 32 KB
  const int tid  = threadIdx.x;
  const int lane = tid & 31;
  const int col  = lane & 15;
  const int h8   = (lane >> 4) * 8;     // A-matrix half select (K packing)
  const int kb   = (lane >> 4) * 16;    // B-matrix half select (K packing)
  const int wave = tid >> 5;
  const int rowbase = blockIdx.x * 128 + wave * 16;

  const unsigned ldsb0 = (unsigned)(size_t)&lds_e[0][0];
  const unsigned ldsb1 = (unsigned)(size_t)&lds_e[1][0];

  // Kick off chunk 0 while we load the A tile from global.
  vq_async_chunk(eb, ldsb0, tid);

  // Load this wave's A tile: 16 rows x 128 K in bf16 (4 x v16bf).
  // 16-bit A layout: lanes 0-15 row M=lane hold K {0..7,16..23}; lanes 16-31
  // hold K {8..15,24..31} of each 32-wide chunk.
  const __bf16* zrow = zb + (size_t)(rowbase + col) * C_DIM;
  v16bf A[4];
#pragma unroll
  for (int q = 0; q < 4; ++q) {
    v8bf lo = *(const v8bf*)(zrow + q * 32 + h8);
    v8bf hi = *(const v8bf*)(zrow + q * 32 + 16 + h8);
    A[q] = __builtin_shufflevector(lo, hi, 0, 1, 2, 3, 4, 5, 6, 7,
                                   8, 9, 10, 11, 12, 13, 14, 15);
  }

  float best[8];
  int bidx[8];
#pragma unroll
  for (int r = 0; r < 8; ++r) { best[r] = 3.4e38f; bidx[r] = 0; }

  asm volatile("s_wait_asynccnt 0x0" ::: "memory");
  __syncthreads();

  for (int cc = 0; cc < K_CODES / 128; ++cc) {
    const __bf16* cur = &lds_e[cc & 1][0];
    // Stream next chunk into the other buffer while computing on this one.
    if (cc < K_CODES / 128 - 1)
      vq_async_chunk(eb + (size_t)(cc + 1) * 128 * C_DIM,
                     (cc & 1) ? ldsb0 : ldsb1, tid);

#pragma unroll 2
    for (int t = 0; t < 8; ++t) {
      // B tile: 16 codes; B layout matches A: K {0..7,16..23} / {8..15,24..31} per half.
      const __bf16* erow = cur + (size_t)(t * 16 + col) * C_DIM;
      const int code = cc * 128 + t * 16 + col;
      const float en = en2[code];          // in flight across the WMMA chain
      // Stage all 4 K-fragments in distinct registers so the 8 ds_load_b128
      // issue back-to-back and the waits stagger across the WMMA chain.
      v16bf Bm[4];
#pragma unroll
      for (int q = 0; q < 4; ++q) {
        v8bf blo = *(const v8bf*)(erow + q * 32 + kb / 2);
        v8bf bhi = *(const v8bf*)(erow + q * 32 + kb / 2 + 16);
        Bm[q] = __builtin_shufflevector(blo, bhi, 0, 1, 2, 3, 4, 5, 6, 7,
                                        8, 9, 10, 11, 12, 13, 14, 15);
      }
      v8f acc = {};
#pragma unroll
      for (int q = 0; q < 4; ++q)
        acc = __builtin_amdgcn_wmma_f32_16x16x32_bf16(
            false, A[q], false, Bm[q], (short)0, acc, false, false);
#pragma unroll
      for (int r = 0; r < 8; ++r) {
        float s = __builtin_fmaf(-2.0f, acc[r], en);
        if (s < best[r]) { best[r] = s; bidx[r] = code; }
      }
    }
    // Next buffer must be fully written (our own async ops) and everyone
    // must be done reading `cur` before it is overwritten next iteration.
    asm volatile("s_wait_asynccnt 0x0" ::: "memory");
    __syncthreads();
  }

  // Reduce over the 16 candidate lanes of each half-wave.
  // C/D layout: VGPR r, lanes 0-15 -> M=r; lanes 16-31 -> M=r+8.
#pragma unroll
  for (int r = 0; r < 8; ++r) {
#pragma unroll
    for (int m = 8; m >= 1; m >>= 1) {
      float ov = __shfl_xor(best[r], m, 32);
      int oi = __shfl_xor(bidx[r], m, 32);
      if (ov < best[r] || (ov == best[r] && oi < bidx[r])) {
        best[r] = ov; bidx[r] = oi;
      }
    }
  }
  if (col == 0) {
    const int mb = rowbase + ((lane >> 4) ? 8 : 0);
#pragma unroll
    for (int r = 0; r < 8; ++r) idx[mb + r] = bidx[r];
  }
}

// ---------------------------------------------------------------------------
// Kernel 4: gather + write quantized (straight-through value == q) + loss.
// out[b][c][hw] = E[idx[b*4096+hw]][c]; accumulate sum((z - q)^2).
// ---------------------------------------------------------------------------
__global__ __launch_bounds__(256) void vq_scatter(const float* __restrict__ z,
                                                  const float* __restrict__ emb,
                                                  const int* __restrict__ idx,
                                                  float* __restrict__ outq,
                                                  double* __restrict__ lossAcc) {
  const int gid = blockIdx.x * 256 + threadIdx.x;   // < 8388608
  const int b   = gid >> 19;          // 128*4096 = 2^19 per batch image
  const int rem = gid & 524287;
  const int c   = rem >> 12;          // 4096 = 2^12
  const int hw  = rem & 4095;
  const int code = idx[(b << 12) + hw];
  const float q  = emb[code * C_DIM + c];
  const float zv = z[gid];
  outq[gid] = q;
  float d = zv - q;
  float p = d * d;
#pragma unroll
  for (int m = 16; m >= 1; m >>= 1) p += __shfl_xor(p, m, 32);
  __shared__ float ws[8];
  if ((threadIdx.x & 31) == 0) ws[threadIdx.x >> 5] = p;
  __syncthreads();
  if (threadIdx.x == 0) {
    float s = 0.f;
#pragma unroll
    for (int i = 0; i < 8; ++i) s += ws[i];
    atomicAdd(lossAcc, (double)s);
  }
}

// ---------------------------------------------------------------------------
// Kernel 5: emit indices (as float, per d_out dtype) and scaled vq_loss.
// vq_loss = (1 + BETA) * mean((z-q)^2), BETA = 0.25 (codebook and commitment
// losses are equal in value; stop_gradient only affects grads).
// ---------------------------------------------------------------------------
__global__ __launch_bounds__(256) void vq_finalize(const int* __restrict__ idx,
                                                   const double* __restrict__ lossAcc,
                                                   float* __restrict__ out) {
  const int gid = blockIdx.x * 256 + threadIdx.x;
  if (gid < N_TOK) out[IDX_OFF + gid] = (float)idx[gid];
  if (gid == 0)
    out[LOSS_OFF] = (float)(lossAcc[0] * (1.25 / ((double)N_TOK * (double)C_DIM)));
}

// ---------------------------------------------------------------------------
extern "C" void kernel_launch(void* const* d_in, const int* in_sizes, int n_in,
                              void* d_out, int out_size, void* d_ws, size_t ws_size,
                              hipStream_t stream) {
  const float* z   = (const float*)d_in[0];   // [16,128,64,64]
  const float* emb = (const float*)d_in[1];   // [1024,128]
  float* out = (float*)d_out;
  char* ws = (char*)d_ws;

  // Workspace layout (16-byte aligned slabs):
  __bf16* zb      = (__bf16*)ws;                                   // 16 MB
  __bf16* eb      = (__bf16*)(ws + 16777216);                      // 256 KB
  float*  en2     = (float*)(ws + 16777216 + 262144);              // 4 KB
  int*    idx     = (int*)(ws + 16777216 + 262144 + 4096);         // 256 KB
  double* lossAcc = (double*)(ws + 16777216 + 262144 + 4096 + 262144);

  vq_zprep<<<dim3(HW / 32, C_DIM / 32, 16), dim3(32, 8), 0, stream>>>(z, zb);
  vq_eprep<<<K_CODES / 8, 256, 0, stream>>>(emb, eb, en2, lossAcc);
  vq_argmin<<<N_TOK / 128, 256, 0, stream>>>(zb, eb, en2, idx);
  vq_scatter<<<OUT_Q / 256, 256, 0, stream>>>(z, emb, idx, out, lossAcc);
  vq_finalize<<<N_TOK / 256, 256, 0, stream>>>(idx, lossAcc, out);
}
